// SeqAE_67284957659692
// MI455X (gfx1250) — matrix operand order repaired and checked by reading, not compiled, not run on hardware
//
#include <hip/hip_runtime.h>

typedef __attribute__((ext_vector_type(16))) __bf16 v16bf;
typedef __attribute__((ext_vector_type(8)))  float  v8f;

#define T_STEPS 800
#define BATCH   256
#define HID     200
#define KP      224               // padded K: 7 chunks of 32; cols 200..203 = [bias, x0, x1, x2]
#define NSEC    208               // padded per-gate width: 13 tiles of 16
#define GROWS   832               // 4 * NSEC padded gate rows
#define SECQ    (NSEC * KP * 2 / 16)  // gate-section stride in uint4 units (5824 -> 93184 B)

union AFrag { uint4 q[2]; v16bf v; };

__device__ __forceinline__ unsigned short f2bf(float f) {
  unsigned int u = __float_as_uint(f);
  u += 0x7FFFu + ((u >> 16) & 1u);          // round-to-nearest-even
  return (unsigned short)(u >> 16);
}
__device__ __forceinline__ float hsig(float z) {
  return fminf(fmaxf(0.2f * z + 0.5f, 0.0f), 1.0f);
}

#if __has_builtin(__builtin_amdgcn_tanhf)
  #define TANHF(x) __builtin_amdgcn_tanhf(x)
#elif __has_builtin(__builtin_amdgcn_tanh_f32)
  #define TANHF(x) __builtin_amdgcn_tanh_f32(x)
#else
  #define TANHF(x) tanhf(x)
#endif

// ---- build extended recurrent matrix [832][224] bf16:
//      cols 0..199  = Whh[g][k]
//      col  200     = bih[g] + bhh[g]
//      cols 201..203 = Wih[g][0..2]
//      everything else (incl. padded gate rows) = 0
__global__ void prep_whh(const float* __restrict__ Whh,
                         const float* __restrict__ Wih,
                         const float* __restrict__ bih,
                         const float* __restrict__ bhh,
                         unsigned short* __restrict__ WhhB) {
  int gid = blockIdx.x * blockDim.x + threadIdx.x;
  if (gid >= GROWS * KP) return;
  int r = gid / KP, k = gid - r * KP;
  int s = r / NSEC, j = r - s * NSEC;
  float v = 0.0f;
  if (j < HID) {
    int g = s * HID + j;
    if (k < HID)           v = Whh[g * HID + k];
    else if (k == HID)     v = bih[g] + bhh[g];
    else if (k <= HID + 3) v = Wih[g * 3 + (k - HID - 1)];
  }
  WhhB[gid] = f2bf(v);
}

// ---- persistent single-workgroup encoder LSTM.
//      h (plus injected [1, x_t] columns) double-buffered in LDS; the WMMA GEMM
//      over K=224 produces the COMPLETE gate pre-activations (recurrent + input
//      + bias), so the fused elementwise is just activations + state update.
__global__ void __launch_bounds__(1024, 1)
enc_lstm(const float* __restrict__ x,            // (256, 3, 800)
         const unsigned short* __restrict__ WhhB,// [832][224] bf16 (L2-resident)
         float* __restrict__ hmaxg,              // [256][200] running max-pool
         float* __restrict__ cws)                // [256][208] cell state
{
  __shared__ unsigned short hlds[2][BATCH * KP]; // 2 x 112 KiB bf16 state

  const int tid = threadIdx.x;
  const int w   = tid >> 5;        // wave 0..31
  const int i0  = w >> 1;          // batch tile 0..15
  const int p   = w & 1;           // j-range half
  const int j0  = p ? 7 : 0;
  const int j1  = p ? 13 : 7;
  const int l   = tid & 31;
  const int m   = l & 15;          // A: M row / B: N col / D: N col
  const int kh  = l >> 4;          // half-wave selector

  for (int i = tid; i < 2 * BATCH * KP; i += 1024) ((unsigned short*)hlds)[i] = 0;
  __syncthreads();

  const int xb = tid >> 2;         // 1024 threads = 256 batches x 4 slots
  const int xf = tid & 3;
  const unsigned int arow = (unsigned)(i0 * 16 + m) * KP + (unsigned)(kh * 16);

  int cur = 0;
  for (int t = 0; t < T_STEPS; ++t) {
    unsigned short* hbw = hlds[cur];
    // inject [1, x0, x1, x2] into K-columns 200..203 of the read buffer
    unsigned short sv = 0x3F80u;                       // bf16 1.0
    if (xf) sv = f2bf(x[(xb * 3 + (xf - 1)) * T_STEPS + t]);
    hbw[xb * KP + HID + xf] = sv;
    __syncthreads();

    const unsigned short* hb = hbw;
    unsigned short*       hx = hlds[cur ^ 1];

    for (int j = j0; j < j1; ++j) {                    // this wave's N-tiles
      v8f acc0 = {}, acc1 = {}, acc2 = {}, acc3 = {};
      // single B base pointer; sections/kc reached via immediate offsets
      const uint4* q = (const uint4*)(WhhB + (unsigned)(j * 16 + m) * KP + (unsigned)(kh * 16));
      #pragma unroll
      for (int kc = 0; kc < 7; ++kc) {
        AFrag a;
        const uint4* ap = (const uint4*)(hb + arow + kc * 32);
        a.q[0] = ap[0];    // K = kh*16 + 0..7
        a.q[1] = ap[1];    // K = kh*16 + 8..15
        AFrag b0, b1, b2, b3;
        b0.q[0] = q[kc * 4 + 0];            b0.q[1] = q[kc * 4 + 1];
        b1.q[0] = q[SECQ     + kc * 4];     b1.q[1] = q[SECQ     + kc * 4 + 1];
        b2.q[0] = q[2 * SECQ + kc * 4];     b2.q[1] = q[2 * SECQ + kc * 4 + 1];
        b3.q[0] = q[3 * SECQ + kc * 4];     b3.q[1] = q[3 * SECQ + kc * 4 + 1];
        acc0 = __builtin_amdgcn_wmma_f32_16x16x32_bf16(false, a.v, false, b0.v, (short)0, acc0, false, false);
        acc1 = __builtin_amdgcn_wmma_f32_16x16x32_bf16(false, a.v, false, b1.v, (short)0, acc1, false, false);
        acc2 = __builtin_amdgcn_wmma_f32_16x16x32_bf16(false, a.v, false, b2.v, (short)0, acc2, false, false);
        acc3 = __builtin_amdgcn_wmma_f32_16x16x32_bf16(false, a.v, false, b3.v, (short)0, acc3, false, false);
      }

      // fused elementwise LSTM update (pre-activations fully in acc)
      const int  n     = j * 16 + m;       // hidden index (D-layout: N = lane&15)
      const bool valid = (n < HID);
      #pragma unroll
      for (int v = 0; v < 8; ++v) {
        int br = i0 * 16 + kh * 8 + v;     // D-layout: M = (lane>>4)*8 + vgpr
        float ig = hsig(acc0[v]);
        float fg = hsig(acc1[v]);
        float og = hsig(acc3[v]);
        float gg = TANHF(acc2[v]);
        int   ci = br * NSEC + n;
        float cp = (t == 0) ? 0.0f : cws[ci];
        float cn = fg * cp + ig * gg;
        cws[ci] = cn;
        float hv = og * TANHF(cn);
        hx[br * KP + n] = valid ? f2bf(hv) : (unsigned short)0;
        if (valid) {
          int hi = br * HID + n;
          hmaxg[hi] = (t == 0) ? hv : fmaxf(hmaxg[hi], hv);
        }
      }
    }
    __syncthreads();
    cur ^= 1;
  }
}

// ---- decoder LSTM (hidden=3): one thread per batch element; the input is the
//      pooled vector repeated over T, so its GEMM contribution is computed once.
__global__ void dec_lstm(const float* __restrict__ hmaxg, // [256][200]
                         const float* __restrict__ Wih,   // (12, 200)
                         const float* __restrict__ Whh,   // (12, 3)
                         const float* __restrict__ bih,
                         const float* __restrict__ bhh,
                         float* __restrict__ out)         // (256, 3, 800)
{
  int b = blockIdx.x * blockDim.x + threadIdx.x;
  if (b >= BATCH) return;
  float pre[12];
  #pragma unroll
  for (int g = 0; g < 12; ++g) pre[g] = bih[g] + bhh[g];
  for (int k = 0; k < HID; ++k) {
    float pv = hmaxg[b * HID + k];
    #pragma unroll
    for (int g = 0; g < 12; ++g) pre[g] += pv * Wih[g * HID + k];
  }
  float whh[12][3];
  #pragma unroll
  for (int g = 0; g < 12; ++g)
    #pragma unroll
    for (int k = 0; k < 3; ++k) whh[g][k] = Whh[g * 3 + k];

  float h[3] = {0.f, 0.f, 0.f}, c[3] = {0.f, 0.f, 0.f};
  for (int t = 0; t < T_STEPS; ++t) {
    float gate[12];
    #pragma unroll
    for (int g = 0; g < 12; ++g)
      gate[g] = pre[g] + whh[g][0] * h[0] + whh[g][1] * h[1] + whh[g][2] * h[2];
    #pragma unroll
    for (int n = 0; n < 3; ++n) {
      float ig = hsig(gate[n]), fg = hsig(gate[3 + n]), og = hsig(gate[9 + n]);
      float gg = TANHF(gate[6 + n]);
      c[n] = fg * c[n] + ig * gg;
      h[n] = og * TANHF(c[n]);
      out[b * 2400 + n * T_STEPS + t] = h[n];
    }
  }
}

extern "C" void kernel_launch(void* const* d_in, const int* in_sizes, int n_in,
                              void* d_out, int out_size, void* d_ws, size_t ws_size,
                              hipStream_t stream) {
  const float* x    = (const float*)d_in[0];
  const float* eWih = (const float*)d_in[1];
  const float* eWhh = (const float*)d_in[2];
  const float* ebih = (const float*)d_in[3];
  const float* ebhh = (const float*)d_in[4];
  const float* dWih = (const float*)d_in[5];
  const float* dWhh = (const float*)d_in[6];
  const float* dbih = (const float*)d_in[7];
  const float* dbhh = (const float*)d_in[8];
  float* out = (float*)d_out;

  // workspace: [bf16 WhhB 832*224 | f32 hmax 256*200 | f32 c 256*208] = ~772 KiB
  unsigned short* WhhB  = (unsigned short*)d_ws;
  float*          hmaxg = (float*)((char*)d_ws + (size_t)GROWS * KP * sizeof(unsigned short));
  float*          cws   = hmaxg + BATCH * HID;

  prep_whh<<<(GROWS * KP + 255) / 256, 256, 0, stream>>>(eWhh, eWih, ebih, ebhh, WhhB);
  enc_lstm<<<1, 1024, 0, stream>>>(x, WhhB, hmaxg, cws);
  dec_lstm<<<2, 128, 0, stream>>>(hmaxg, dWih, dWhh, dbih, dbhh, out);
}
